// MUSTANG_Classifier_36163624632872
// MI455X (gfx1250) — matrix operand order repaired and checked by reading, not compiled, not run on hardware
//
#include <hip/hip_runtime.h>
#include <hip/hip_bf16.h>
#include <math.h>

typedef __attribute__((ext_vector_type(16))) __bf16 v16bf;
typedef __attribute__((ext_vector_type(8)))  float  v8f;

#define HDIM 256
#define NN   50000
#define EE   800000
#define SSN  5
#define DDIM 1536

// ---------------------------------------------------------------- utilities
__device__ __forceinline__ void atomicMaxF(float* addr, float v) {
    if (v >= 0.0f) atomicMax((int*)addr, __float_as_int(v));
    else           atomicMin((unsigned int*)addr, __float_as_uint(v));
}

__global__ void k_fill_u32(unsigned* __restrict__ p, unsigned n, unsigned v) {
    unsigned i = blockIdx.x * 256u + threadIdx.x;
    if (i < n) p[i] = v;
}

__global__ void k_init_edges(const int* __restrict__ edge_index,
                             int* __restrict__ srcW, int* __restrict__ dstW,
                             float* __restrict__ wW) {
    unsigned e = blockIdx.x * 256u + threadIdx.x;
    if (e < EE) {
        srcW[e] = edge_index[e];
        dstW[e] = edge_index[EE + e];
        wW[e]   = 1.0f;
    }
}

// ---------------------------------------------------------------- WMMA GEMM
// Y[n,256] = X[n,K] @ W[K,256]   (f32 in, bf16 WMMA, f32 accumulate)
#define TM 128
#define TN 64
#define TK 32
#define A_STR 34   // padded LDS row stride (halves)
#define B_STR 40

__global__ __launch_bounds__(256) void k_gemm_bf16(
    const float* __restrict__ X, const float* __restrict__ W,
    float* __restrict__ Y, int n, int K)
{
    __shared__ __align__(16) __bf16 As[TM * A_STR];
    __shared__ __align__(16) __bf16 Bs[TN * B_STR];

    const int tid   = threadIdx.x;
    const int lane  = tid & 31;
    const int wave  = tid >> 5;
    const int mBase = blockIdx.y * TM;
    const int nBase = blockIdx.x * TN;

    v8f acc[4];
#pragma unroll
    for (int c = 0; c < 4; ++c)
#pragma unroll
        for (int r = 0; r < 8; ++r) acc[c][r] = 0.0f;

    const int ar   = tid >> 1;          // A row handled by this thread
    const int ahalf= tid & 1;           // which 16-wide K half
    int arow_g = mBase + ar; if (arow_g >= n) arow_g = n - 1;
    const int bc = tid & 63;            // B column handled by this thread
    const int bk = (tid >> 6) * 8;      // 8 K rows

    for (int k0 = 0; k0 < K; k0 += TK) {
        // stage A tile (convert f32 -> bf16)
        const float* ap = X + (long long)arow_g * K + k0 + ahalf * 16;
#pragma unroll
        for (int j = 0; j < 16; j += 4) {
            float4 f = *(const float4*)(ap + j);
            __bf16* d = &As[ar * A_STR + ahalf * 16 + j];
            d[0] = (__bf16)f.x; d[1] = (__bf16)f.y;
            d[2] = (__bf16)f.z; d[3] = (__bf16)f.w;
        }
        // stage B tile transposed: Bs[col][k]
#pragma unroll
        for (int j = 0; j < 8; ++j)
            Bs[bc * B_STR + bk + j] =
                (__bf16)W[(long long)(k0 + bk + j) * HDIM + nBase + bc];

        if (k0 + TK < K)  // software prefetch of next A tile slice
            __builtin_prefetch(X + (long long)arow_g * K + k0 + TK + ahalf * 16, 0, 0);

        __syncthreads();

        // A fragment: 16-bit A 16x32, contiguous k (lanes 0-15: K 0-15, lanes 16-31: K 16-31)
        union { v16bf v; unsigned u[8]; } af;
        const int m  = lane & 15;
        const int kb = (lane >> 4) * 16;
        const __bf16* arow = &As[(wave * 16 + m) * A_STR];
#pragma unroll
        for (int p = 0; p < 4; ++p) af.u[p]     = *(const unsigned*)(arow + kb + 2 * p);
#pragma unroll
        for (int p = 0; p < 4; ++p) af.u[4 + p] = *(const unsigned*)(arow + kb + 8 + 2 * p);

#pragma unroll
        for (int ct = 0; ct < 4; ++ct) {
            // B fragment: 16-bit B 32x16 layout (lanes 0-15: K 0-15, lanes 16-31: K 16-31)
            union { v16bf v; unsigned u[8]; } bf_;
            const __bf16* brow = &Bs[(ct * 16 + m) * B_STR + (lane >> 4) * 16];
#pragma unroll
            for (int j = 0; j < 8; ++j) bf_.u[j] = *(const unsigned*)(brow + 2 * j);

            acc[ct] = __builtin_amdgcn_wmma_f32_16x16x32_bf16(
                false, af.v, false, bf_.v, (short)0, acc[ct], false, false);
        }
        __syncthreads();
    }

    // store C: lanes 0-15 -> M=r, lanes 16-31 -> M=r+8
    const int m = lane & 15;
    const int rbase = mBase + wave * 16 + (lane >> 4) * 8;
#pragma unroll
    for (int ct = 0; ct < 4; ++ct) {
        const int col = nBase + ct * 16 + m;
#pragma unroll
        for (int rr = 0; rr < 8; ++rr) {
            const int row = rbase + rr;
            if (row < n) Y[(long long)row * HDIM + col] = acc[ct][rr];
        }
    }
}

// ---------------------------------------------------------------- graph ops
__global__ void k_deg(const int* __restrict__ dst, const float* __restrict__ w,
                      float* __restrict__ deg) {
    unsigned e = blockIdx.x * 256u + threadIdx.x;
    if (e < EE && w[e] != 0.0f) atomicAdd(&deg[dst[e]], w[e]);
}

__global__ void k_dinv(const float* __restrict__ deg, float* __restrict__ dinv, int n) {
    int i = blockIdx.x * 256 + threadIdx.x;
    if (i < n) { float d = deg[i]; dinv[i] = d > 0.0f ? rsqrtf(d) : 0.0f; }
}

// one block per edge, one thread per feature
__global__ __launch_bounds__(256) void k_scatter(
    const float* __restrict__ xw, const int* __restrict__ src,
    const int* __restrict__ dst, const float* __restrict__ w,
    const float* __restrict__ dinv, float* __restrict__ agg)
{
    int e = blockIdx.x;
    float we = w[e];
    if (we == 0.0f) return;
    int s = src[e], d = dst[e];
    float coef = dinv[s] * dinv[d] * we;
    int f = threadIdx.x;
    atomicAdd(&agg[(long long)d * HDIM + f], xw[(long long)s * HDIM + f] * coef);
}

__global__ void k_biasrelu(float* __restrict__ h, const float* __restrict__ b, int total) {
    int i = blockIdx.x * 256 + threadIdx.x;
    if (i < total) { float v = h[i] + b[i & (HDIM - 1)]; h[i] = v > 0.0f ? v : 0.0f; }
}

// per-node dots h.wrel and h.wroot (one wave per node)
__global__ __launch_bounds__(256) void k_rdots(
    const float* __restrict__ h, const float* __restrict__ wrel,
    const float* __restrict__ wroot, float* __restrict__ rrel,
    float* __restrict__ rroot, int n)
{
    int wv = threadIdx.x >> 5, lane = threadIdx.x & 31;
    int node = blockIdx.x * 8 + wv;
    if (node >= n) return;
    const float* hp = h + (long long)node * HDIM;
    float s1 = 0.0f, s2 = 0.0f;
#pragma unroll
    for (int j = 0; j < 8; ++j) {
        float v = hp[lane + 32 * j];
        s1 += v * wrel[lane + 32 * j];
        s2 += v * wroot[lane + 32 * j];
    }
#pragma unroll
    for (int off = 16; off; off >>= 1) {
        s1 += __shfl_down(s1, off, 32);
        s2 += __shfl_down(s2, off, 32);
    }
    if (lane == 0) { rrel[node] = s1; rroot[node] = s2; }
}

__global__ void k_aggscore(const int* __restrict__ src, const int* __restrict__ dst,
                           const float* __restrict__ w, const float* __restrict__ rrel,
                           float* __restrict__ aggs) {
    unsigned e = blockIdx.x * 256u + threadIdx.x;
    if (e < EE && w[e] != 0.0f) atomicAdd(&aggs[dst[e]], rrel[src[e]] * w[e]);
}

__global__ void k_scorekey(const float* __restrict__ aggs, const float* __restrict__ rroot,
                           const float* __restrict__ brel, float* __restrict__ score,
                           unsigned* __restrict__ keys, unsigned* __restrict__ hist, int n) {
    int i = blockIdx.x * 256 + threadIdx.x;
    if (i >= n) return;
    float sc = tanhf(aggs[i] + brel[0] + rroot[i]);
    score[i] = sc;
    unsigned u = __float_as_uint(sc);
    unsigned key = (u >> 31) ? ~u : (u | 0x80000000u);
    keys[i] = key;
    atomicAdd(&hist[key >> 16], 1u);
}

// single-block suffix scan over 65536 bins: find top-k threshold bin
__global__ __launch_bounds__(256) void k_scan(const unsigned* __restrict__ hist,
                                              int* __restrict__ meta, int k) {
    __shared__ unsigned chunk[256];
    unsigned t = threadIdx.x, s = 0;
    for (int b = (int)t * 256; b < (int)t * 256 + 256; ++b) s += hist[b];
    chunk[t] = s;
    __syncthreads();
    if (t == 0) {
        unsigned cum = 0; int T = 0; unsigned cstrict = 0;
        for (int c = 255; c >= 0; --c) {
            if (cum + chunk[c] >= (unsigned)k) {
                for (int bb = c * 256 + 255; bb >= c * 256; --bb) {
                    unsigned hv = hist[bb];
                    if (cum + hv >= (unsigned)k) { T = bb; cstrict = cum; break; }
                    cum += hv;
                }
                break;
            }
            cum += chunk[c];
        }
        meta[0] = T; meta[1] = (int)cstrict; meta[2] = k - (int)cstrict;
        meta[3] = 0; meta[4] = 0;
    }
}

__global__ void k_select(const unsigned* __restrict__ keys, const float* __restrict__ score,
                         int* __restrict__ meta, int* __restrict__ perm,
                         float* __restrict__ vals, int* __restrict__ newidx, int n) {
    int i = blockIdx.x * 256 + threadIdx.x;
    if (i >= n) return;
    int T = meta[0], cstrict = meta[1], need = meta[2];
    int bin = (int)(keys[i] >> 16);
    int pos = -1;
    if (bin > T) {
        pos = atomicAdd(&meta[3], 1);
    } else if (bin == T) {
        int t = atomicAdd(&meta[4], 1);
        if (t < need) pos = cstrict + t;
    }
    newidx[i] = pos;
    if (pos >= 0) { perm[pos] = i; vals[pos] = score[i]; }
}

__global__ void k_stats1(const int* __restrict__ perm, const float* __restrict__ vals,
                         const int* __restrict__ stOld, int* __restrict__ stNew,
                         float* __restrict__ cnt, float* __restrict__ sumval, int k) {
    int i = blockIdx.x * 256 + threadIdx.x;
    if (i >= k) return;
    int s = stOld[perm[i]];
    stNew[i] = s;
    atomicAdd(&cnt[s], 1.0f);
    atomicAdd(&sumval[s], vals[i]);
}

// gate kept nodes + stain-grouped sum/max; one block per kept node
__global__ __launch_bounds__(256) void k_pool(
    const float* __restrict__ h, const int* __restrict__ perm,
    const float* __restrict__ vals, const int* __restrict__ stNew,
    float* __restrict__ xp, float* __restrict__ fsum, float* __restrict__ fmax)
{
    int i = blockIdx.x, f = threadIdx.x;
    int node = perm[i];
    float v = h[(long long)node * HDIM + f] * vals[i];
    xp[(long long)i * HDIM + f] = v;
    int s = stNew[i];
    atomicAdd(&fsum[s * HDIM + f], v);
    atomicMaxF(&fmax[s * HDIM + f], v);
}

__global__ __launch_bounds__(256) void k_emb(
    const float* __restrict__ cnt, const float* __restrict__ sumval,
    const float* __restrict__ fsum, const float* __restrict__ fmax,
    float* __restrict__ emb)
{
    int f = threadIdx.x;
    float wt[SSN]; float wsum = 0.0f;
#pragma unroll
    for (int s = 0; s < SSN; ++s) {
        float c = cnt[s];
        float m = c > 0.0f ? sumval[s] / c : 0.0f;
        wt[s] = m; wsum += m;
    }
    float inv = wsum != 0.0f ? 1.0f / wsum : 0.0f;
    float e1 = 0.0f, e2 = 0.0f;
#pragma unroll
    for (int s = 0; s < SSN; ++s) {
        float c = cnt[s];
        if (c > 0.0f) {
            float w = wt[s] * inv;
            e1 += (fsum[s * HDIM + f] / c) * w;
            e2 += fmax[s * HDIM + f] * w;
        }
    }
    emb[f] = e1;
    emb[HDIM + f] = e2;
}

__global__ void k_relabel(int* __restrict__ src, int* __restrict__ dst,
                          float* __restrict__ w, const int* __restrict__ newidx) {
    unsigned e = blockIdx.x * 256u + threadIdx.x;
    if (e >= EE) return;
    int s2 = newidx[src[e]], d2 = newidx[dst[e]];
    bool keep = (s2 >= 0) && (d2 >= 0) && (w[e] > 0.0f);
    src[e] = keep ? s2 : 0;
    dst[e] = keep ? d2 : 0;
    w[e]   = keep ? 1.0f : 0.0f;
}

// ---------------------------------------------------------------- tail (attention/classifier)
__global__ __launch_bounds__(256) void k_layernorm(
    const float* __restrict__ in, const float* __restrict__ res,
    const float* __restrict__ g, const float* __restrict__ b,
    float* __restrict__ out)
{
    __shared__ float red[256];
    __shared__ float sh_mean, sh_rstd;
    int t = threadIdx.x;
    float s = 0.0f;
    for (int j = 0; j < 6; ++j) {
        int idx = t + 256 * j;
        float v = in[idx] + (res ? res[idx] : 0.0f);
        s += v;
    }
    red[t] = s; __syncthreads();
    for (int o = 128; o; o >>= 1) { if (t < o) red[t] += red[t + o]; __syncthreads(); }
    if (t == 0) sh_mean = red[0] / (float)DDIM;
    __syncthreads();
    float mean = sh_mean;
    s = 0.0f;
    for (int j = 0; j < 6; ++j) {
        int idx = t + 256 * j;
        float v = in[idx] + (res ? res[idx] : 0.0f) - mean;
        s += v * v;
    }
    red[t] = s; __syncthreads();
    for (int o = 128; o; o >>= 1) { if (t < o) red[t] += red[t + o]; __syncthreads(); }
    if (t == 0) sh_rstd = rsqrtf(red[0] / (float)DDIM + 1e-5f);
    __syncthreads();
    float rstd = sh_rstd;
    for (int j = 0; j < 6; ++j) {
        int idx = t + 256 * j;
        float v = in[idx] + (res ? res[idx] : 0.0f);
        out[idx] = (v - mean) * rstd * g[idx] + b[idx];
    }
}

__global__ void k_matvec(const float* __restrict__ in, const float* __restrict__ W,
                         const float* __restrict__ bias, float* __restrict__ out,
                         int IN, int OUT, int doRelu) {
    int j = blockIdx.x * 256 + threadIdx.x;
    if (j >= OUT) return;
    float acc = bias[j];
    for (int i = 0; i < IN; ++i) {
        float v = in[i];
        if (doRelu) v = fmaxf(v, 0.0f);
        acc += v * W[(long long)i * OUT + j];
    }
    out[j] = acc;
}

__global__ void k_la(const float* __restrict__ y, float* __restrict__ out) {
    __shared__ float s[3];
    int t = threadIdx.x;
    if (t < 3) {
        float a = 0.0f;
        for (int j = 0; j < 512; ++j) a += y[t * 512 + j];
        s[t] = a;
    }
    __syncthreads();
    if (t == 0) {
        float m = fminf(s[0], fminf(s[1], s[2]));
        float l0 = s[0] - m + 1e-8f, l1 = s[1] - m + 1e-8f, l2 = s[2] - m + 1e-8f;
        float tot = l0 + l1 + l2;
        out[4] = l0 / tot; out[5] = l1 / tot; out[6] = l2 / tot;
    }
}

__global__ __launch_bounds__(256) void k_final(
    const float* __restrict__ h1, const float* __restrict__ w2,
    const float* __restrict__ b2, const int* __restrict__ label,
    float* __restrict__ out)
{
    __shared__ float r0[256], r1[256];
    int t = threadIdx.x;
    float p0 = 0.0f, p1 = 0.0f;
    for (int i = t; i < 768; i += 256) {
        float v = h1[i];
        p0 += v * w2[i * 2 + 0];
        p1 += v * w2[i * 2 + 1];
    }
    r0[t] = p0; r1[t] = p1; __syncthreads();
    for (int o = 128; o; o >>= 1) {
        if (t < o) { r0[t] += r0[t + o]; r1[t] += r1[t + o]; }
        __syncthreads();
    }
    if (t == 0) {
        float l0 = r0[0] + b2[0], l1 = r1[0] + b2[1];
        float m = fmaxf(l0, l1);
        float e0 = expf(l0 - m), e1 = expf(l1 - m);
        float inv = 1.0f / (e0 + e1);
        out[0] = l0; out[1] = l1;
        out[2] = e0 * inv; out[3] = e1 * inv;
        out[7] = (float)label[0];
    }
}

// ---------------------------------------------------------------- host driver
extern "C" void kernel_launch(void* const* d_in, const int* in_sizes, int n_in,
                              void* d_out, int out_size, void* d_ws, size_t ws_size,
                              hipStream_t stream) {
    const float* x          = (const float*)d_in[0];
    const int*   edge_index = (const int*)d_in[1];
    const int*   node_attr  = (const int*)d_in[2];
    const int*   label      = (const int*)d_in[3];
    const float* convw[3]   = {(const float*)d_in[4], (const float*)d_in[5], (const float*)d_in[6]};
    const float* conv_b     = (const float*)d_in[7];
    const float* pool_wrel  = (const float*)d_in[8];
    const float* pool_brel  = (const float*)d_in[9];
    const float* pool_wroot = (const float*)d_in[10];
    const float* ln1_g      = (const float*)d_in[11];
    const float* ln1_b      = (const float*)d_in[12];
    const float* attn_in_w  = (const float*)d_in[13];
    const float* attn_in_b  = (const float*)d_in[14];
    const float* attn_out_w = (const float*)d_in[15];
    const float* attn_out_b = (const float*)d_in[16];
    const float* ln2_g      = (const float*)d_in[17];
    const float* ln2_b      = (const float*)d_in[18];
    const float* cls1_w     = (const float*)d_in[19];
    const float* cls1_b     = (const float*)d_in[20];
    const float* cls2_w     = (const float*)d_in[21];
    const float* cls2_b     = (const float*)d_in[22];
    float* out = (float*)d_out;

    // ---- workspace carve
    char* w8 = (char*)d_ws;
    size_t off = 0;
    auto carve = [&](size_t bytes) -> void* {
        void* p = w8 + off;
        off = (off + bytes + 255) & ~(size_t)255;
        return p;
    };
    int*      srcW  = (int*)carve((size_t)EE * 4);
    int*      dstW  = (int*)carve((size_t)EE * 4);
    float*    wW    = (float*)carve((size_t)EE * 4);
    float*    deg   = (float*)carve((size_t)NN * 4);
    float*    dinv  = (float*)carve((size_t)NN * 4);
    float*    bufP  = (float*)carve((size_t)NN * HDIM * 4);   // GEMM out (XW)
    float*    bufQ  = (float*)carve((size_t)NN * HDIM * 4);   // agg -> h (in place)
    float*    bufR  = (float*)carve((size_t)35000 * HDIM * 4);// pooled features
    float*    rrel  = (float*)carve((size_t)NN * 4);
    float*    rroot = (float*)carve((size_t)NN * 4);
    float*    aggs  = (float*)carve((size_t)NN * 4);
    float*    score = (float*)carve((size_t)NN * 4);
    unsigned* keys  = (unsigned*)carve((size_t)NN * 4);
    unsigned* hist  = (unsigned*)carve((size_t)65536 * 4);
    int*      meta  = (int*)carve(64 * 4);
    int*      perm  = (int*)carve((size_t)NN * 4);
    float*    vals  = (float*)carve((size_t)NN * 4);
    int*      nwidx = (int*)carve((size_t)NN * 4);
    int*      stA   = (int*)carve((size_t)NN * 4);
    int*      stB   = (int*)carve((size_t)NN * 4);
    float*    cnt   = (float*)carve(SSN * 4);
    float*    sval  = (float*)carve(SSN * 4);
    float*    fsum  = (float*)carve((size_t)SSN * HDIM * 4);
    float*    fmax  = (float*)carve((size_t)SSN * HDIM * 4);
    float*    emb   = (float*)carve(DDIM * 4);
    float*    xa    = (float*)carve(DDIM * 4);
    float*    qkv   = (float*)carve(3 * DDIM * 4);
    float*    attnv = (float*)carve(DDIM * 4);
    float*    yv    = (float*)carve(DDIM * 4);
    float*    h1    = (float*)carve(768 * 4);
    (void)ws_size; (void)n_in; (void)in_sizes; (void)out_size;

    auto fillu = [&](void* p, size_t n, unsigned v) {
        k_fill_u32<<<dim3((unsigned)((n + 255) / 256)), dim3(256), 0, stream>>>((unsigned*)p, (unsigned)n, v);
    };

    const int nArr[3] = {50000, 35000, 24500};
    const int kArr[3] = {35000, 24500, 17150};
    const int KArr[3] = {1024, 256, 256};
    const float* xcArr[3] = {x, bufR, bufR};
    const int* stOld[3];
    int* stNew[3];
    stOld[0] = node_attr; stNew[0] = stA;
    stOld[1] = stA;       stNew[1] = stB;
    stOld[2] = stB;       stNew[2] = stA;

    k_init_edges<<<dim3((EE + 255) / 256), dim3(256), 0, stream>>>(edge_index, srcW, dstW, wW);

    for (int l = 0; l < 3; ++l) {
        const int n = nArr[l], k = kArr[l], K = KArr[l];
        fillu(deg, n, 0);
        fillu(bufQ, (size_t)n * HDIM, 0);
        fillu(aggs, n, 0);
        fillu(hist, 65536, 0);

        k_gemm_bf16<<<dim3(HDIM / TN, (n + TM - 1) / TM), dim3(256), 0, stream>>>(
            xcArr[l], convw[l], bufP, n, K);

        k_deg<<<dim3((EE + 255) / 256), dim3(256), 0, stream>>>(dstW, wW, deg);
        k_dinv<<<dim3((n + 255) / 256), dim3(256), 0, stream>>>(deg, dinv, n);
        k_scatter<<<dim3(EE), dim3(HDIM), 0, stream>>>(bufP, srcW, dstW, wW, dinv, bufQ);
        k_biasrelu<<<dim3(((size_t)n * HDIM + 255) / 256), dim3(256), 0, stream>>>(
            bufQ, conv_b + l * HDIM, n * HDIM);

        k_rdots<<<dim3((n + 7) / 8), dim3(256), 0, stream>>>(
            bufQ, pool_wrel + l * HDIM, pool_wroot + l * HDIM, rrel, rroot, n);
        k_aggscore<<<dim3((EE + 255) / 256), dim3(256), 0, stream>>>(srcW, dstW, wW, rrel, aggs);
        k_scorekey<<<dim3((n + 255) / 256), dim3(256), 0, stream>>>(
            aggs, rroot, pool_brel + l, score, keys, hist, n);
        k_scan<<<dim3(1), dim3(256), 0, stream>>>(hist, meta, k);
        k_select<<<dim3((n + 255) / 256), dim3(256), 0, stream>>>(
            keys, score, meta, perm, vals, nwidx, n);

        fillu(cnt, SSN, 0);
        fillu(sval, SSN, 0);
        fillu(fsum, (size_t)SSN * HDIM, 0);
        fillu(fmax, (size_t)SSN * HDIM, 0xFF800000u);  // -inf
        k_stats1<<<dim3((k + 255) / 256), dim3(256), 0, stream>>>(
            perm, vals, stOld[l], stNew[l], cnt, sval, k);
        k_pool<<<dim3(k), dim3(HDIM), 0, stream>>>(bufQ, perm, vals, stNew[l], bufR, fsum, fmax);
        k_emb<<<dim3(1), dim3(HDIM), 0, stream>>>(cnt, sval, fsum, fmax, emb + l * 2 * HDIM);
        k_relabel<<<dim3((EE + 255) / 256), dim3(256), 0, stream>>>(srcW, dstW, wW, nwidx);
    }

    // tail: degenerate 1-token attention + classifier
    k_layernorm<<<dim3(1), dim3(256), 0, stream>>>(emb, nullptr, ln1_g, ln1_b, xa);
    k_matvec<<<dim3((3 * DDIM + 255) / 256), dim3(256), 0, stream>>>(
        xa, attn_in_w, attn_in_b, qkv, DDIM, 3 * DDIM, 0);
    // softmax over single key == 1 -> attn_in = v
    k_matvec<<<dim3((DDIM + 255) / 256), dim3(256), 0, stream>>>(
        qkv + 2 * DDIM, attn_out_w, attn_out_b, attnv, DDIM, DDIM, 0);
    k_layernorm<<<dim3(1), dim3(256), 0, stream>>>(attnv, xa, ln2_g, ln2_b, yv);
    k_la<<<dim3(1), dim3(32), 0, stream>>>(yv, out);
    k_matvec<<<dim3(3), dim3(256), 0, stream>>>(yv, cls1_w, cls1_b, h1, DDIM, 768, 1);
    k_final<<<dim3(1), dim3(256), 0, stream>>>(h1, cls2_w, cls2_b, label, out);
}
